// LightWinModel_21036749815833
// MI455X (gfx1250) — matrix operand order repaired and checked by reading, not compiled, not run on hardware
//
#include <hip/hip_runtime.h>

typedef __attribute__((ext_vector_type(16))) _Float16 v16h;
typedef __attribute__((ext_vector_type(8)))  float    v8f;

#define N_BUCKETS 50
#define HALF      16
#define HIDDEN    32

// One wave (32 lanes) processes tiles of 16 samples.
// Layer1 (K=2) is computed by VALU straight into WMMA A-matrix layout.
// Layer2 ([16,32]x[32,16]) is a single v_wmma_f32_16x16x32_f16 per tile.
// Head gather + dot is done on the D registers with a 16-lane xor reduction.
__global__ __launch_bounds__(256) void lightwin_wmma_kernel(
    const float* __restrict__ x, const int* __restrict__ buckets,
    const float* __restrict__ W1, const float* __restrict__ b1,
    const float* __restrict__ W2, const float* __restrict__ b2,
    const float* __restrict__ Wh, const float* __restrict__ bh,
    float* __restrict__ out, int nTiles)
{
    const int lane          = threadIdx.x & 31;
    const int waveInBlock   = threadIdx.x >> 5;
    const int wavesPerBlock = blockDim.x >> 5;
    const int globalWave    = blockIdx.x * wavesPerBlock + waveInBlock;
    const int nWaves        = gridDim.x * wavesPerBlock;

    const int  m      = lane & 15;       // A: row (sample-in-tile); B/D: column N
    const bool hiHalf = lane >= 16;
    const int  jbase  = hiHalf ? 8 : 0;  // layer1 output-channel base for this half

    // ---- loop-invariant layer1 weights, sliced per lane's A K-slots ----
    // slot s<8  -> j = jbase + s          (K =  0..7  / 8..15)
    // slot s>=8 -> j = 16 + jbase + (s-8) (K = 16..23 / 24..31)
    float w0[16], w1[16], bb[16];
    #pragma unroll
    for (int s = 0; s < 16; ++s) {
        int j = (s < 8) ? (jbase + s) : (16 + jbase + (s - 8));
        w0[s] = W1[0 * HIDDEN + j];
        w1[s] = W1[1 * HIDDEN + j];
        bb[s] = b1[j];
    }

    // ---- W2 [32,16] row-major -> B-matrix layout (column = m, K over slots) ----
    v16h bmat;
    const int kb = jbase;                // lanes 0-15: K=0..7,16..23, lanes 16-31: K=8..15,24..31
    #pragma unroll
    for (int k = 0; k < 16; ++k)
        bmat[k] = (_Float16)W2[(k < 8 ? kb + k : 16 + kb + (k - 8)) * HALF + m];

    const float cbias = b2[m];           // C/D column N = lane%16

    for (int tile = globalWave; tile < nTiles; tile += nWaves) {
        const int base = tile * 16;

        // both halves load the same 16 float2s (coalesced, L0-hit for upper half)
        const float2 xv = *(const float2*)(x + 2 * (base + m));
        __builtin_prefetch(x + 2 * (base + nWaves * 16 + m), 0, 3);

        // ---- layer1: relu(x@W1 + b1), produced directly in A layout ----
        v16h a;
        #pragma unroll
        for (int s = 0; s < 16; ++s) {
            float h = fmaf(xv.x, w0[s], fmaf(xv.y, w1[s], bb[s]));
            a[s] = (_Float16)fmaxf(h, 0.0f);
        }

        // ---- layer2: D = A(16x32) * B(32x16) + b2 ----
        v8f c = {cbias, cbias, cbias, cbias, cbias, cbias, cbias, cbias};
        c = __builtin_amdgcn_wmma_f32_16x16x32_f16(
                /*neg_a=*/false, a, /*neg_b=*/false, bmat,
                /*c_mod=*/(short)0, c, /*reuse_a=*/false, /*reuse_b=*/false);

        // ---- per-bucket head: gather Wh row, dot along N (across 16 lanes) ----
        int bid = buckets[base + m];
        int id  = bid < N_BUCKETS ? bid : (N_BUCKETS - 1);

        float r[8];
        #pragma unroll
        for (int mm = 0; mm < 8; ++mm) {
            // sample handled by c[mm]: M = mm (lower half) / mm+8 (upper half);
            // its bucket id lives in lane jbase+mm.
            int   idm  = __shfl(id, jbase + mm, 32);
            float part = c[mm] * Wh[idm * HALF + m];
            part += __shfl_xor(part, 1, 32);
            part += __shfl_xor(part, 2, 32);
            part += __shfl_xor(part, 4, 32);
            part += __shfl_xor(part, 8, 32);   // stays within each 16-lane half
            r[mm] = part + bh[idm];
        }

        if (m == 0) {                          // lanes 0 and 16 write 8 outputs each
            float4 lo = {r[0], r[1], r[2], r[3]};
            float4 hi = {r[4], r[5], r[6], r[7]};
            float* o = out + base + (hiHalf ? 8 : 0);
            *(float4*)(o + 0) = lo;
            *(float4*)(o + 4) = hi;
        }
    }
}

// Scalar tail for B % 16 leftovers (no WMMA: EXEC may be partial).
__global__ void lightwin_tail_kernel(
    const float* __restrict__ x, const int* __restrict__ buckets,
    const float* __restrict__ W1, const float* __restrict__ b1,
    const float* __restrict__ W2, const float* __restrict__ b2,
    const float* __restrict__ Wh, const float* __restrict__ bh,
    float* __restrict__ out, int start, int Btot)
{
    int i = start + blockIdx.x * blockDim.x + threadIdx.x;
    if (i >= Btot) return;
    float x0 = x[2 * i + 0], x1 = x[2 * i + 1];
    float h1[HIDDEN];
    #pragma unroll
    for (int j = 0; j < HIDDEN; ++j)
        h1[j] = fmaxf(fmaf(x0, W1[j], fmaf(x1, W1[HIDDEN + j], b1[j])), 0.0f);
    int bid = buckets[i];
    int id  = bid < N_BUCKETS ? bid : (N_BUCKETS - 1);
    float acc = 0.0f;
    #pragma unroll
    for (int n = 0; n < HALF; ++n) {
        float h = b2[n];
        #pragma unroll
        for (int j = 0; j < HIDDEN; ++j)
            h = fmaf(h1[j], W2[j * HALF + n], h);
        acc = fmaf(h, Wh[id * HALF + n], acc);
    }
    out[i] = acc + bh[id];
}

extern "C" void kernel_launch(void* const* d_in, const int* in_sizes, int n_in,
                              void* d_out, int out_size, void* d_ws, size_t ws_size,
                              hipStream_t stream) {
    const float* x       = (const float*)d_in[0];
    const int*   buckets = (const int*)  d_in[1];
    const float* W1      = (const float*)d_in[2];
    const float* b1      = (const float*)d_in[3];
    const float* W2      = (const float*)d_in[4];
    const float* b2      = (const float*)d_in[5];
    const float* Wh      = (const float*)d_in[6];
    const float* bh      = (const float*)d_in[7];
    float*       out     = (float*)d_out;

    const int B      = in_sizes[0] / 2;   // x is [B,2]
    const int nTiles = B / 16;
    const int rem    = B - nTiles * 16;

    if (nTiles > 0) {
        int blocks = (nTiles + 7) / 8;    // 8 waves per block
        if (blocks > 1024) blocks = 1024; // 8192 waves grid-striding 125k tiles
        lightwin_wmma_kernel<<<blocks, 256, 0, stream>>>(
            x, buckets, W1, b1, W2, b2, Wh, bh, out, nTiles);
    }
    if (rem > 0) {
        lightwin_tail_kernel<<<1, 64, 0, stream>>>(
            x, buckets, W1, b1, W2, b2, Wh, bh, out, nTiles * 16, B);
    }
}
